// AdditiveAttention_3229815407167
// MI455X (gfx1250) — matrix operand order repaired and checked by reading, not compiled, not run on hardware
//
#include <hip/hip_runtime.h>
#include <hip/hip_bf16.h>
#include <math.h>

// ---------------------------------------------------------------------------
// AdditiveAttention for MI455X (gfx1250, wave32, WMMA bf16 path)
// B=64, S=2048, HID=1024, ATTN=512
// ---------------------------------------------------------------------------

#define BATCH 64
#define SEQ   2048
#define HID   1024
#define ATTN  512

typedef __bf16 bf16_t;
typedef __attribute__((ext_vector_type(16))) __bf16 v16bf;
typedef __attribute__((ext_vector_type(8)))  __bf16 v8bf;
typedef __attribute__((ext_vector_type(4)))  __bf16 v4bf;
typedef __attribute__((ext_vector_type(8)))  float  v8f;

// Padded LDS row: 1024 bf16 + 8 pad elements (16B) -> 2064B stride,
// bank stride 4 dwords => lanes 0-15 hit disjoint bank quads on ds_load_b128.
#define ROWPAD 1032

#define LOG2E  1.4426950408889634f

// Branch-free tanh: TRANS32 exp2 + rcp co-execute with the WMMA/XDL pipe.
// tanh(x) = 1 - 2/(exp(2x)+1); exact at +-inf limits (rcp(inf)=0).
__device__ __forceinline__ float fast_tanh(float x) {
    float e = __builtin_amdgcn_exp2f(x * (2.0f * LOG2E));  // exp(2x)
    float r = __builtin_amdgcn_rcpf(e + 1.0f);
    return fmaf(-2.0f, r, 1.0f);
}

// ---------------------------------------------------------------------------
// Kernel 1: convert W1 [ATTN,HID] f32 -> bf16 once (2MB -> 1MB, L2-resident)
// ---------------------------------------------------------------------------
__global__ __launch_bounds__(256) void cvt_w1_kernel(const float* __restrict__ W1,
                                                     bf16_t* __restrict__ w1bf) {
    int idx = blockIdx.x * 256 + threadIdx.x;          // handles 4 elements
    float4 f = ((const float4*)W1)[idx];
    v4bf o;
    o[0] = (bf16_t)f.x; o[1] = (bf16_t)f.y; o[2] = (bf16_t)f.z; o[3] = (bf16_t)f.w;
    ((v4bf*)w1bf)[idx] = o;
}

// ---------------------------------------------------------------------------
// Kernel 2: proj_dec[b][n] = dot(dec_hidden[b,:], W2[n,:])   (tiny, fp32)
// ---------------------------------------------------------------------------
__global__ __launch_bounds__(256) void pdec_kernel(const float* __restrict__ dec,
                                                   const float* __restrict__ W2,
                                                   float* __restrict__ pdec) {
    int idx = blockIdx.x * 256 + threadIdx.x;          // 0..B*ATTN-1
    int b = idx >> 9;                                  // /ATTN
    int n = idx & (ATTN - 1);
    const float4* d4 = (const float4*)(dec + (long)b * HID);
    const float4* w4 = (const float4*)(W2 + (long)n * HID);
    float acc = 0.f;
    #pragma unroll 8
    for (int i = 0; i < HID / 4; ++i) {
        float4 a = d4[i], w = w4[i];
        acc += a.x * w.x + a.y * w.y + a.z * w.z + a.w * w.w;
    }
    pdec[idx] = acc;
}

// ---------------------------------------------------------------------------
// Kernel 3 (the big one): fused GEMM + tanh + v-dot + mask -> scores[B,S]
// Each wave owns 16 rows of M=B*S. 4 waves per WG => 64 rows per WG.
// A (enc rows) staged in LDS as bf16 once; proj_enc never materialized.
// ---------------------------------------------------------------------------
__global__ __launch_bounds__(128) void energy_kernel(
        const float*  __restrict__ enc,     // [B,S,HID] f32
        const bf16_t* __restrict__ w1bf,    // [ATTN,HID] bf16
        const float*  __restrict__ pdec,    // [B,ATTN]
        const float*  __restrict__ vvec,    // [ATTN]
        const int*    __restrict__ mask,    // [B,S]
        float*        __restrict__ scores)  // [B,S]
{
    extern __shared__ bf16_t smem[];                   // 4 * 16 * ROWPAD bf16
    const int lane = threadIdx.x & 31;
    const int wave = threadIdx.x >> 5;
    const int lh = lane & 15;                          // column / row selector
    const int hi = lane >> 4;                          // half-wave id (0/1)

    const long Rbase = (long)blockIdx.x * 64 + (long)wave * 16; // first M row
    const int  b  = (int)(Rbase >> 11);                // Rbase / SEQ
    const int  s0 = (int)(Rbase & (SEQ - 1));

    bf16_t* aTile = smem + wave * 16 * ROWPAD;

    // ---- Stage A: 16 rows x 1024 f32 -> bf16 into LDS (read enc once) ----
    const float* src = enc + Rbase * HID;
    #pragma unroll 4
    for (int it = 0; it < 64; ++it) {
        int t   = it * 32 + lane;                      // chunk id 0..2047
        int row = t >> 7;                              // 128 chunks / row
        int k   = (t & 127) * 8;
        const float4* g = (const float4*)(src + (long)row * HID + k);
        float4 f0 = g[0], f1 = g[1];
        v8bf o;
        o[0] = (bf16_t)f0.x; o[1] = (bf16_t)f0.y;
        o[2] = (bf16_t)f0.z; o[3] = (bf16_t)f0.w;
        o[4] = (bf16_t)f1.x; o[5] = (bf16_t)f1.y;
        o[6] = (bf16_t)f1.z; o[7] = (bf16_t)f1.w;
        *(v8bf*)(aTile + row * ROWPAD + k) = o;
    }
    __syncthreads();

    float sc[8];
    #pragma unroll
    for (int r = 0; r < 8; ++r) sc[r] = 0.f;

    // A fragment base: row = lh, K offset hi*16 (lanes>=16 take K+16, matching B)
    const bf16_t* aBase = aTile + lh * ROWPAD + hi * 16;

    // ---- Loop over N tiles (32 x 16 columns) ----
    for (int nt = 0; nt < 32; ++nt) {
        const int n = nt * 16 + lh;                    // this lane's column
        // B fragment: 16 contiguous K values of W1[n,:] at K = kt*32 + hi*16
        const bf16_t* bBase = w1bf + (long)n * HID + hi * 16;

        v8f acc = {0.f, 0.f, 0.f, 0.f, 0.f, 0.f, 0.f, 0.f};
        #pragma unroll 4
        for (int kt = 0; kt < 32; ++kt) {
            // A frag: two 16B LDS loads -> K [+0..7] and [+8..15] (hi-shifted)
            v8bf alo = *(const v8bf*)(aBase + kt * 32);
            v8bf ahi = *(const v8bf*)(aBase + kt * 32 + 8);
            v16bf afrag = __builtin_shufflevector(alo, ahi,
                0, 1, 2, 3, 4, 5, 6, 7, 8, 9, 10, 11, 12, 13, 14, 15);
            // B frag: one contiguous 32B global load (hits L2/WGP$)
            v16bf bfrag = *(const v16bf*)(bBase + kt * 32);
            acc = __builtin_amdgcn_wmma_f32_16x16x32_bf16(
                false, afrag, false, bfrag, (short)0, acc, false, false);
        }
        // Epilogue for this 16x16 tile:
        // acc[r] = C[M = r + 8*hi][N = lh];  fold tanh + v-dot immediately.
        const float pd = pdec[b * ATTN + n];
        const float vv = vvec[n];
        #pragma unroll
        for (int r = 0; r < 8; ++r)
            sc[r] += vv * fast_tanh(acc[r] + pd);
    }

    // ---- Cross-lane reduce over the 16 columns in each half-wave ----
    #pragma unroll
    for (int r = 0; r < 8; ++r) {
        float s = sc[r];
        s += __shfl_xor(s, 1, 32);
        s += __shfl_xor(s, 2, 32);
        s += __shfl_xor(s, 4, 32);
        s += __shfl_xor(s, 8, 32);
        sc[r] = s;
    }
    if (lh == 0) {                                     // lanes 0 and 16 write
        #pragma unroll
        for (int r = 0; r < 8; ++r) {
            int m = r + hi * 8;
            int s = s0 + m;
            float val = sc[r];
            if (mask[(long)b * SEQ + s] == 0) val = -1e9f;
            scores[(long)b * SEQ + s] = val;
        }
    }
}

// ---------------------------------------------------------------------------
// Kernel 4: softmax over S per batch row (deterministic tree reductions)
// ---------------------------------------------------------------------------
__global__ __launch_bounds__(256) void softmax_kernel(const float* __restrict__ scores,
                                                      float* __restrict__ attn) {
    __shared__ float red[256];
    const int b = blockIdx.x;
    const int t = threadIdx.x;
    float v[8];
    float mx = -INFINITY;
    #pragma unroll
    for (int i = 0; i < 8; ++i) {
        v[i] = scores[(long)b * SEQ + t + i * 256];
        mx = fmaxf(mx, v[i]);
    }
    red[t] = mx; __syncthreads();
    for (int off = 128; off > 0; off >>= 1) {
        if (t < off) red[t] = fmaxf(red[t], red[t + off]);
        __syncthreads();
    }
    mx = red[0]; __syncthreads();

    float sum = 0.f;
    #pragma unroll
    for (int i = 0; i < 8; ++i) {
        // args are <= 0 after max subtraction: exp2 underflows cleanly to 0
        v[i] = __builtin_amdgcn_exp2f((v[i] - mx) * LOG2E);
        sum += v[i];
    }
    red[t] = sum; __syncthreads();
    for (int off = 128; off > 0; off >>= 1) {
        if (t < off) red[t] += red[t + off];
        __syncthreads();
    }
    sum = red[0];
    const float inv = 1.f / sum;
    #pragma unroll
    for (int i = 0; i < 8; ++i)
        attn[(long)b * SEQ + t + i * 256] = v[i] * inv;
}

// ---------------------------------------------------------------------------
// Kernel 5: context partials  (split S 16 ways for bandwidth; no atomics)
// part[sp][b][h] = sum_{s in slab} attn[b,s] * enc[b,s,h]
// ---------------------------------------------------------------------------
__global__ __launch_bounds__(256) void ctx_partial_kernel(const float* __restrict__ attn,
                                                          const float* __restrict__ enc,
                                                          float* __restrict__ part) {
    __shared__ float aslab[SEQ / 16];                  // 128 attn weights
    const int b  = blockIdx.x;
    const int sp = blockIdx.y;
    const int h  = threadIdx.x * 4;
    if (threadIdx.x < SEQ / 16)
        aslab[threadIdx.x] = attn[(long)b * SEQ + sp * (SEQ / 16) + threadIdx.x];
    __syncthreads();

    float4 acc = {0.f, 0.f, 0.f, 0.f};
    #pragma unroll 4
    for (int i = 0; i < SEQ / 16; ++i) {
        int s = sp * (SEQ / 16) + i;
        float a = aslab[i];
        float4 e = *(const float4*)(enc + ((long)b * SEQ + s) * HID + h);
        acc.x += a * e.x; acc.y += a * e.y; acc.z += a * e.z; acc.w += a * e.w;
    }
    *(float4*)(part + ((long)sp * BATCH + b) * HID + h) = acc;
}

__global__ __launch_bounds__(256) void ctx_reduce_kernel(const float* __restrict__ part,
                                                         float* __restrict__ ctx) {
    int idx = blockIdx.x * 256 + threadIdx.x;          // 0 .. B*HID-1
    float s = 0.f;
    #pragma unroll
    for (int sp = 0; sp < 16; ++sp)
        s += part[(long)sp * (BATCH * HID) + idx];
    ctx[idx] = s;
}

// ---------------------------------------------------------------------------
// Host launcher
// ---------------------------------------------------------------------------
extern "C" void kernel_launch(void* const* d_in, const int* in_sizes, int n_in,
                              void* d_out, int out_size, void* d_ws, size_t ws_size,
                              hipStream_t stream) {
    (void)in_sizes; (void)n_in; (void)out_size; (void)ws_size;

    const float* dec  = (const float*)d_in[0];   // [B,HID]
    const float* enc  = (const float*)d_in[1];   // [B,S,HID]
    const int*   mask = (const int*)  d_in[2];   // [B,S]
    const float* W1   = (const float*)d_in[3];   // [ATTN,HID]
    const float* W2   = (const float*)d_in[4];   // [ATTN,HID]
    const float* vvec = (const float*)d_in[5];   // [1,ATTN]

    float* ctx_out  = (float*)d_out;                       // [B,HID]
    float* attn_out = (float*)d_out + BATCH * HID;         // [B,S]

    // Workspace layout (bytes):
    char* ws = (char*)d_ws;
    float*  pdec   = (float*) (ws + 0);                    // B*ATTN*4    = 128KB
    bf16_t* w1bf   = (bf16_t*)(ws + 131072);               // ATTN*HID*2  = 1MB
    float*  scores = (float*) (ws + 131072 + 1048576);     // B*S*4       = 512KB
    float*  part   = (float*) (ws + 131072 + 1048576 + 524288); // 16*B*HID*4 = 4MB

    // 1) W1 -> bf16
    cvt_w1_kernel<<<(ATTN * HID / 4) / 256, 256, 0, stream>>>(W1, w1bf);
    // 2) proj_dec
    pdec_kernel<<<(BATCH * ATTN) / 256, 256, 0, stream>>>(dec, W2, pdec);
    // 3) fused GEMM+tanh+v-dot+mask -> scores (WMMA bf16)
    {
        const size_t ldsBytes = 4u * 16u * ROWPAD * sizeof(bf16_t); // 132096
        energy_kernel<<<(BATCH * SEQ) / 64, 128, ldsBytes, stream>>>(
            enc, w1bf, pdec, vvec, mask, scores);
    }
    // 4) softmax -> attn output
    softmax_kernel<<<BATCH, 256, 0, stream>>>(scores, attn_out);
    // 5) context = attn @ enc (split-S partials, then deterministic reduce)
    ctx_partial_kernel<<<dim3(BATCH, 16), 256, 0, stream>>>(attn_out, enc, part);
    ctx_reduce_kernel<<<(BATCH * HID) / 256, 256, 0, stream>>>(part, ctx_out);
}
